// TrajectoryEncoder_34574486733330
// MI455X (gfx1250) — compile-verified, matrix operand order checked by reading, not run on hardware
//
#include <hip/hip_runtime.h>
#include <hip/hip_bf16.h>

// ---------------------------------------------------------------------------
// TrajectoryEncoder fused kernel for gfx1250 (MI455X).
// One workgroup (8 wave32) per polyline; all intermediates live in LDS.
// GEMMs use v_wmma_f32_16x16x32_f16 (f16 in, f32 accumulate); BatchNorm gamma
// and 1/sqrt(1+eps) folded into f16 weights by a one-shot convert kernel.
// ---------------------------------------------------------------------------

typedef __attribute__((ext_vector_type(16))) _Float16 v16h;
typedef __attribute__((ext_vector_type(8)))  _Float16 v8h;
typedef __attribute__((ext_vector_type(8)))  float    v8f;

#define LDA 520   // halves; 1040B row stride -> fragment rows hit distinct banks
#define LDB 264   // halves; 528B row stride

// workspace layout (halves)
#define OFF_W0 0u          // pre_w0  256x32
#define OFF_W1 8192u       // pre_w1  256x256
#define OFF_W2 73728u      // pre_w2  256x256
#define OFF_M0 139264u     // mlp_w0  256x512
#define OFF_M1 270336u     // mlp_w1  256x256
#define OFF_M2 335872u     // mlp_w2  256x256
#define OFF_O0 401408u     // out_w0  256x256
#define OFF_O1 466944u     // out_w1  256x256

// f32 -> f16 weight conversion with per-row BN scale folded in.
__global__ void convert_w_kernel(const float* __restrict__ src,
                                 _Float16* __restrict__ dst,
                                 int n, int cols,
                                 const float* __restrict__ gamma, float scl) {
    int i = blockIdx.x * blockDim.x + threadIdx.x;
    if (i >= n) return;
    int r = i / cols;
    float s = scl * (gamma ? gamma[r] : 1.0f);
    dst[i] = (_Float16)(src[i] * s);
}

// A fragment: 16x32 f16, rows from LDS (row-major, stride ld halves).
// ISA layout: lane L -> row (L&15); halves h: K = 8*(L>>4) + (h&7) + 16*(h>>3).
__device__ __forceinline__ v16h ldfragA(const _Float16* base, int ld,
                                        int mrow, int kk, int lane) {
    int row = mrow + (lane & 15);
    int kb  = kk + ((lane >> 4) << 3);
    union { v16h v; v8h h[2]; } u;
    const _Float16* p = base + row * ld + kb;
    u.h[0] = *(const v8h*)(p);
    u.h[1] = *(const v8h*)(p + 16);
    return u.v;
}

// B fragment: 32x16 f16 from global weights stored as W[n][k] row-major
// (so B[k][n] = W[n][k], i.e. exactly x @ W.T). Same per-lane K pattern,
// lane indexes the output column.
__device__ __forceinline__ v16h ldfragB(const _Float16* w, int K,
                                        int ncol, int kk, int lane) {
    int col = ncol + (lane & 15);
    int kb  = kk + ((lane >> 4) << 3);
    union { v16h v; v8h h[2]; } u;
    const _Float16* p = w + (size_t)col * K + kb;
    u.h[0] = *(const v8h*)(p);
    u.h[1] = *(const v8h*)(p + 16);
    return u.v;
}

// One dense layer: sOut[32x256] = epilogue(sIn[32xK] @ W.T + bias).
// sMaskRow == nullptr -> ReLU epilogue; else multiply row mask (no ReLU).
__device__ __forceinline__ void gemm_layer(const _Float16* __restrict__ sIn, int ldin, int K,
                                           const _Float16* __restrict__ Wh,
                                           const float* __restrict__ bias,
                                           _Float16* __restrict__ sOut, int ldout,
                                           const float* __restrict__ sMaskRow,
                                           int lane, int wave) {
    const int n0 = wave * 32;                 // 8 waves x 32 cols = 256 cols
    __builtin_prefetch((const void*)(Wh + (size_t)n0 * K), 0, 1);
    v8f acc00 = {}, acc01 = {}, acc10 = {}, acc11 = {};
    for (int kk = 0; kk < K; kk += 32) {
        v16h a0 = ldfragA(sIn, ldin, 0,  kk, lane);
        v16h a1 = ldfragA(sIn, ldin, 16, kk, lane);
        v16h b0 = ldfragB(Wh, K, n0,      kk, lane);
        v16h b1 = ldfragB(Wh, K, n0 + 16, kk, lane);
        acc00 = __builtin_amdgcn_wmma_f32_16x16x32_f16(false, a0, false, b0, (short)0, acc00, false, false);
        acc01 = __builtin_amdgcn_wmma_f32_16x16x32_f16(false, a0, false, b1, (short)0, acc01, false, false);
        acc10 = __builtin_amdgcn_wmma_f32_16x16x32_f16(false, a1, false, b0, (short)0, acc10, false, false);
        acc11 = __builtin_amdgcn_wmma_f32_16x16x32_f16(false, a1, false, b1, (short)0, acc11, false, false);
    }
    // C/D layout: vgpr r, lane L -> M = r + 8*(L>>4), N = L&15.
    const int rowoff = (lane >> 4) << 3;
    const int c0 = n0 + (lane & 15);
    const int c1 = c0 + 16;
    const float bb0 = bias[c0], bb1 = bias[c1];
#pragma unroll
    for (int r = 0; r < 8; ++r) {
        int m0 = r + rowoff, m1 = 16 + r + rowoff;
        float v00 = acc00[r] + bb0, v01 = acc01[r] + bb1;
        float v10 = acc10[r] + bb0, v11 = acc11[r] + bb1;
        if (sMaskRow) {
            float mk0 = sMaskRow[m0], mk1 = sMaskRow[m1];
            v00 *= mk0; v01 *= mk0; v10 *= mk1; v11 *= mk1;
        } else {
            v00 = fmaxf(v00, 0.f); v01 = fmaxf(v01, 0.f);
            v10 = fmaxf(v10, 0.f); v11 = fmaxf(v11, 0.f);
        }
        sOut[m0 * ldout + c0] = (_Float16)v00;
        sOut[m0 * ldout + c1] = (_Float16)v01;
        sOut[m1 * ldout + c0] = (_Float16)v10;
        sOut[m1 * ldout + c1] = (_Float16)v11;
    }
}

__global__ __launch_bounds__(256)
void polyline_encoder_kernel(const float* __restrict__ poly,
                             const unsigned char* __restrict__ mask,
                             const _Float16* __restrict__ W,
                             const float* __restrict__ pre_b0,
                             const float* __restrict__ pre_b1,
                             const float* __restrict__ pre_bias2,
                             const float* __restrict__ mlp_b0,
                             const float* __restrict__ mlp_b1,
                             const float* __restrict__ mlp_bias2,
                             const float* __restrict__ out_b0,
                             const float* __restrict__ out_b1,
                             float* __restrict__ out) {
    __shared__ __align__(16) _Float16 sA[32 * LDA];   // up to 32 x 512 activations
    __shared__ __align__(16) _Float16 sB[32 * LDB];   // 32 x 256 activations
    __shared__ float sMask[32];
    __shared__ float sBuf[256];
    __shared__ float sY[256];
    __shared__ float sValid;

    const int tid  = threadIdx.x;
    const int lane = tid & 31;
    const int wave = tid >> 5;
    const int bp   = blockIdx.x;                      // polyline index in [0, B*P)

    // ---- stage input: 20x32 f32 -> f16 in sA (rows 20..31 zero-padded) ----
    const float* xin = poly + (size_t)bp * (20 * 32);
    for (int idx = tid; idx < 32 * 32; idx += 256) {
        int r = idx >> 5, c = idx & 31;
        float v = (r < 20) ? xin[r * 32 + c] : 0.0f;
        sA[r * LDA + c] = (_Float16)v;
    }
    if (tid < 32) sMask[tid] = (tid < 20 && mask[(size_t)bp * 20 + tid]) ? 1.0f : 0.0f;
    if (tid == 0) {
        float any = 0.0f;
        for (int r = 0; r < 20; ++r) if (mask[(size_t)bp * 20 + r]) any = 1.0f;
        sValid = any;
    }
    __syncthreads();

    // ---- pre MLP: [Lin+BN+ReLU] x2, Lin+bias, then mask-zero ----
    gemm_layer(sA, LDA, 32,  W + OFF_W0, pre_b0,    sB, LDB, nullptr, lane, wave);
    __syncthreads();
    gemm_layer(sB, LDB, 256, W + OFF_W1, pre_b1,    sA, LDA, nullptr, lane, wave);
    __syncthreads();
    gemm_layer(sA, LDA, 256, W + OFF_W2, pre_bias2, sB, LDB, sMask,   lane, wave);
    __syncthreads();

    // ---- maxpool over the 20 points + concat [x, pooled] into sA (32x512) ----
    {
        int t = tid;                                  // one column per thread
        float mx = (float)sB[t];
        for (int r = 1; r < 20; ++r) mx = fmaxf(mx, (float)sB[r * LDB + t]);
        _Float16 pool = (_Float16)mx;
        for (int r = 0; r < 32; ++r) {
            sA[r * LDA + t]       = sB[r * LDB + t];  // x
            sA[r * LDA + 256 + t] = pool;             // pooled broadcast
        }
    }
    __syncthreads();

    // ---- mlps: 2H -> H (BN+ReLU), H -> H (BN+ReLU), H -> H (+bias, mask) ----
    gemm_layer(sA, LDA, 512, W + OFF_M0, mlp_b0,    sB, LDB, nullptr, lane, wave);
    __syncthreads();
    gemm_layer(sB, LDB, 256, W + OFF_M1, mlp_b1,    sA, LDA, nullptr, lane, wave);
    __syncthreads();
    gemm_layer(sA, LDA, 256, W + OFF_M2, mlp_bias2, sB, LDB, sMask,   lane, wave);
    __syncthreads();

    // ---- final maxpool -> per-polyline feature buf[256] (f32) ----
    {
        int t = tid;
        float mx = (float)sB[t];
        for (int r = 1; r < 20; ++r) mx = fmaxf(mx, (float)sB[r * LDB + t]);
        sBuf[t] = mx;
    }
    __syncthreads();

    // ---- out_mlps (M=1: VALU dot products, f16 weights from L2) ----
    {
        int n = tid;
        const _Float16* w = W + OFF_O0 + (size_t)n * 256;
        float acc = out_b0[n];
        for (int k = 0; k < 256; k += 8) {
            v8h wv = *(const v8h*)(w + k);
#pragma unroll
            for (int j = 0; j < 8; ++j) acc += (float)wv[j] * sBuf[k + j];
        }
        sY[n] = fmaxf(acc, 0.0f);
    }
    __syncthreads();
    {
        int n = tid;
        const _Float16* w = W + OFF_O1 + (size_t)n * 256;
        float acc = out_b1[n];
        for (int k = 0; k < 256; k += 8) {
            v8h wv = *(const v8h*)(w + k);
#pragma unroll
            for (int j = 0; j < 8; ++j) acc += (float)wv[j] * sY[k + j];
        }
        out[(size_t)bp * 256 + n] = acc * sValid;
    }
}

extern "C" void kernel_launch(void* const* d_in, const int* in_sizes, int n_in,
                              void* d_out, int out_size, void* d_ws, size_t ws_size,
                              hipStream_t stream) {
    const float*         poly      = (const float*)d_in[0];
    const unsigned char* mask      = (const unsigned char*)d_in[1];
    const float* pre_w0 = (const float*)d_in[2];
    const float* pre_g0 = (const float*)d_in[3];
    const float* pre_b0 = (const float*)d_in[4];
    const float* pre_w1 = (const float*)d_in[5];
    const float* pre_g1 = (const float*)d_in[6];
    const float* pre_b1 = (const float*)d_in[7];
    const float* pre_w2 = (const float*)d_in[8];
    const float* pre_bias2 = (const float*)d_in[9];
    const float* mlp_w0 = (const float*)d_in[10];
    const float* mlp_g0 = (const float*)d_in[11];
    const float* mlp_b0 = (const float*)d_in[12];
    const float* mlp_w1 = (const float*)d_in[13];
    const float* mlp_g1 = (const float*)d_in[14];
    const float* mlp_b1 = (const float*)d_in[15];
    const float* mlp_w2 = (const float*)d_in[16];
    const float* mlp_bias2 = (const float*)d_in[17];
    const float* out_w0 = (const float*)d_in[18];
    const float* out_b0 = (const float*)d_in[19];
    const float* out_w1 = (const float*)d_in[20];
    const float* out_b1 = (const float*)d_in[21];

    _Float16* W = (_Float16*)d_ws;
    const float BN_INV = 0.99999500003749969f;  // 1/sqrt(1 + 1e-5)

    struct Cvt { const float* src; unsigned off; int rows, cols; const float* g; };
    const Cvt cv[8] = {
        { pre_w0, OFF_W0, 256,  32, pre_g0 },
        { pre_w1, OFF_W1, 256, 256, pre_g1 },
        { pre_w2, OFF_W2, 256, 256, nullptr },
        { mlp_w0, OFF_M0, 256, 512, mlp_g0 },
        { mlp_w1, OFF_M1, 256, 256, mlp_g1 },
        { mlp_w2, OFF_M2, 256, 256, nullptr },
        { out_w0, OFF_O0, 256, 256, nullptr },
        { out_w1, OFF_O1, 256, 256, nullptr },
    };
    for (int i = 0; i < 8; ++i) {
        int n = cv[i].rows * cv[i].cols;
        convert_w_kernel<<<(n + 255) / 256, 256, 0, stream>>>(
            cv[i].src, W + cv[i].off, n, cv[i].cols, cv[i].g,
            cv[i].g ? BN_INV : 1.0f);
    }

    const int BP = 8 * 1024;  // B * P polylines, one workgroup each
    polyline_encoder_kernel<<<BP, 256, 0, stream>>>(
        poly, mask, W,
        pre_b0, pre_b1, pre_bias2,
        mlp_b0, mlp_b1, mlp_bias2,
        out_b0, out_b1,
        (float*)d_out);
}
